// CausalSelfAttention_47150150976164
// MI455X (gfx1250) — matrix operand order repaired and checked by reading, not compiled, not run on hardware
//
#include <hip/hip_runtime.h>
#include <hip/hip_bf16.h>
#include <stdint.h>

// ---------------------------------------------------------------------------
// Causal self-attention for MI455X (gfx1250, wave32, WMMA bf16, async-to-LDS)
//   B=4, T=2048, C=1024, H=16, HD=64
// Pipeline:
//   0) convert x -> bf16 ; transpose w_qkv, w_proj -> bf16 [N][K]
//   1) gemm (EPI=1): qk = x@w_qkv+b (bf16 [M][2C]), v -> transposed [B][HD*H][T]
//   2) flash attention (WMMA 16x16x32 bf16, online softmax)   -> bf16 [M][C]
//   3) gemm (EPI=0): out = att@w_proj + b_proj                -> f32 d_out
// All LDS tile staging uses global_load_async_to_lds_b128 when available.
// ---------------------------------------------------------------------------

typedef __bf16 bf16_t;
typedef bf16_t v16bf __attribute__((ext_vector_type(16)));
typedef bf16_t v8bf  __attribute__((ext_vector_type(8)));
typedef float  v8f   __attribute__((ext_vector_type(8)));
typedef int    v4i   __attribute__((ext_vector_type(4)));

#define CSA_B  4
#define CSA_T  2048
#define CSA_C  1024
#define CSA_H  16
#define CSA_HD 64

#if defined(__gfx1250__) && __has_builtin(__builtin_amdgcn_global_load_async_to_lds_b128)
#define CSA_ASYNC 1
#else
#define CSA_ASYNC 0
#endif

typedef __attribute__((address_space(1))) v4i as1_v4i;   // global
typedef __attribute__((address_space(3))) v4i as3_v4i;   // LDS

// 16-byte global -> LDS copy (per lane). Async path uses the CDNA5
// ASYNCcnt-tracked direct-to-LDS instruction; fallback is a v8bf copy.
static __device__ __forceinline__ void copy16_g2l(void* lds, const void* g) {
#if CSA_ASYNC
    __builtin_amdgcn_global_load_async_to_lds_b128(
        (as1_v4i*)(uintptr_t)g, (as3_v4i*)(uintptr_t)lds, 0, 0);
#else
    *(v8bf*)lds = *(const v8bf*)g;
#endif
}

static __device__ __forceinline__ void copy_wait() {
#if CSA_ASYNC
#if __has_builtin(__builtin_amdgcn_s_wait_asynccnt)
    __builtin_amdgcn_s_wait_asynccnt(0);
#else
    asm volatile("s_wait_asynccnt 0x0" ::: "memory");
#endif
#endif
}

static __device__ __forceinline__ v16bf pack16(v8bf lo, v8bf hi) {
    v16bf r;
#pragma unroll
    for (int e = 0; e < 8; ++e) { r[e] = lo[e]; r[8 + e] = hi[e]; }
    return r;
}

// ---------------------------------------------------------------------------
// 0a) fp32 -> bf16 elementwise (8 elems / thread)
// ---------------------------------------------------------------------------
__global__ __launch_bounds__(256) void convert_f32_bf16_kernel(
    const float* __restrict__ x, bf16_t* __restrict__ y)
{
    const size_t i = ((size_t)blockIdx.x * blockDim.x + threadIdx.x) * 8;
    float4 a = *(const float4*)(x + i);
    float4 b = *(const float4*)(x + i + 4);
    v8bf o;
    o[0] = (bf16_t)a.x; o[1] = (bf16_t)a.y; o[2] = (bf16_t)a.z; o[3] = (bf16_t)a.w;
    o[4] = (bf16_t)b.x; o[5] = (bf16_t)b.y; o[6] = (bf16_t)b.z; o[7] = (bf16_t)b.w;
    *(v8bf*)(y + i) = o;
}

// ---------------------------------------------------------------------------
// 0b) fp32 [K][N] -> bf16 transposed [N][K] via LDS 32x32 tile
// ---------------------------------------------------------------------------
__global__ __launch_bounds__(256) void transpose_f32_bf16_kernel(
    const float* __restrict__ w, bf16_t* __restrict__ wt, int K, int N)
{
    __shared__ float tile[32][33];
    const int n0 = blockIdx.x * 32, k0 = blockIdx.y * 32;
    const int tx = threadIdx.x, ty = threadIdx.y;   // 32 x 8
#pragma unroll
    for (int i = 0; i < 32; i += 8)
        tile[ty + i][tx] = w[(size_t)(k0 + ty + i) * N + n0 + tx];
    __syncthreads();
#pragma unroll
    for (int i = 0; i < 32; i += 8)
        wt[(size_t)(n0 + ty + i) * K + k0 + tx] = (bf16_t)tile[tx][ty + i];
}

// ---------------------------------------------------------------------------
// GEMM: out[M,N] = A[M,K](bf16) @ Bt[N,K](bf16)^T + bias
// Block 128x128, BK=32, 256 threads = 8 waves (4 along M x 2 along N),
// wave tile 32x64 = 2x4 WMMA accumulators. Both LDS tiles contiguous-along-K,
// staged with per-lane async b128 copies.
// EPI=0: plain TOut row-major. EPI=1: QKV split (q,k -> qk[M][2C]; v -> vT).
// ---------------------------------------------------------------------------
template <typename TOut, int EPI>
__global__ __launch_bounds__(256) void gemm_bt_kernel(
    const bf16_t* __restrict__ A,    // [M][K]
    const bf16_t* __restrict__ Bt,   // [N][K]
    const float*  __restrict__ bias, // [N]
    TOut* __restrict__ Cout,         // EPI==0
    bf16_t* __restrict__ qk,         // EPI==1: [M][2C]
    bf16_t* __restrict__ vT,         // EPI==1: [B][H*HD][T]
    int M, int N, int K)
{
    constexpr int BM = 128, BN = 128, BK = 32, PAD = 8;
    __shared__ alignas(16) bf16_t sA [BM][BK + PAD];   // [m][k]
    __shared__ alignas(16) bf16_t sBt[BN][BK + PAD];   // [n][k]

    const int tid    = threadIdx.x;
    const int lane   = tid & 31;
    const int wid    = tid >> 5;
    const int wave_m = wid & 3;     // 32 rows each
    const int wave_n = wid >> 2;    // 64 cols each
    const int half   = lane >> 4;
    const int l16    = lane & 15;

    const int blk_n = blockIdx.x * BN;
    const int blk_m = blockIdx.y * BM;

    v8f acc[2][4];
#pragma unroll
    for (int mt = 0; mt < 2; ++mt)
#pragma unroll
        for (int nt = 0; nt < 4; ++nt) acc[mt][nt] = (v8f)0.0f;

    for (int k0 = 0; k0 < K; k0 += BK) {
        // stage both tiles: 128 rows x 64 B each side = 1024 x 16B units
#pragma unroll
        for (int i = 0; i < 2; ++i) {
            const int u = tid + i * 256;          // 0..511
            const int row = u >> 2, seg = u & 3;  // 4 x 16B per 64B row
            copy16_g2l(&sA[row][seg * 8],
                       A  + (size_t)(blk_m + row) * K + k0 + seg * 8);
            copy16_g2l(&sBt[row][seg * 8],
                       Bt + (size_t)(blk_n + row) * K + k0 + seg * 8);
        }
        copy_wait();
        __syncthreads();

        // A fragment: a[e] = A[m][(e<8 ? e : e+8) + half*8]
        v16bf af[2];
#pragma unroll
        for (int mt = 0; mt < 2; ++mt) {
            const int row = wave_m * 32 + mt * 16 + l16;
            v8bf lo = *(const v8bf*)&sA[row][half * 8];
            v8bf hi = *(const v8bf*)&sA[row][16 + half * 8];
            af[mt] = pack16(lo, hi);
        }
        // B fragment: b[e] = B[(e<8 ? e : e+8) + half*8][n] with sBt = [n][k]
        v16bf bfg[4];
#pragma unroll
        for (int nt = 0; nt < 4; ++nt) {
            const int row = wave_n * 64 + nt * 16 + l16;
            v8bf lo = *(const v8bf*)&sBt[row][half * 8];
            v8bf hi = *(const v8bf*)&sBt[row][16 + half * 8];
            bfg[nt] = pack16(lo, hi);
        }
#pragma unroll
        for (int mt = 0; mt < 2; ++mt)
#pragma unroll
            for (int nt = 0; nt < 4; ++nt)
                acc[mt][nt] = __builtin_amdgcn_wmma_f32_16x16x32_bf16(
                    false, af[mt], false, bfg[nt], (short)0, acc[mt][nt],
                    false, false);
        __syncthreads();
    }

    // epilogue: C layout row = r + half*8 (VGPR r), col = l16
#pragma unroll
    for (int mt = 0; mt < 2; ++mt)
#pragma unroll
        for (int nt = 0; nt < 4; ++nt) {
            const int col  = blk_n + wave_n * 64 + nt * 16 + l16;
            const float bv = bias[col];
            const int row0 = blk_m + wave_m * 32 + mt * 16 + half * 8;
            if constexpr (EPI == 1) {
                if (col < 2 * CSA_C) {   // q,k -> [M][2C]
#pragma unroll
                    for (int r = 0; r < 8; ++r)
                        qk[(size_t)(row0 + r) * (2 * CSA_C) + col] =
                            (bf16_t)(acc[mt][nt][r] + bv);
                } else {                 // v -> vT[B][H*HD][T], 8 consecutive t
                    const int c2   = col - 2 * CSA_C;   // h*HD + hd
                    const int bidx = row0 / CSA_T;
                    const int t0   = row0 % CSA_T;
                    v8bf pk;
#pragma unroll
                    for (int r = 0; r < 8; ++r)
                        pk[r] = (bf16_t)(acc[mt][nt][r] + bv);
                    *(v8bf*)&vT[((size_t)bidx * CSA_H * CSA_HD + c2) * CSA_T + t0] = pk;
                }
            } else {
#pragma unroll
                for (int r = 0; r < 8; ++r)
                    Cout[(size_t)(row0 + r) * N + col] =
                        (TOut)(acc[mt][nt][r] + bv);
            }
        }
}

// ---------------------------------------------------------------------------
// Flash attention. Grid: (T/64, B*H). 128 threads = 4 waves; each wave owns
// 16 query rows. K staged [key][hd], V staged pre-transposed [hd][key]; both
// via per-lane async b128 copies (no in-kernel transposes).
// ---------------------------------------------------------------------------
__global__ __launch_bounds__(128) void attn_kernel(
    const bf16_t* __restrict__ qk,    // [B*T][2C]  (q | k)
    const bf16_t* __restrict__ vT,    // [B][H*HD][T]
    bf16_t* __restrict__ attout)      // [B*T][C]
{
    constexpr int T = CSA_T, C = CSA_C, H = CSA_H, HD = CSA_HD;
    constexpr int C2 = 2 * C;

    __shared__ alignas(16) bf16_t sK [32][HD + 8];    // [key][hd], stride 72
    __shared__ alignas(16) bf16_t sVt[HD][32 + 8];    // [hd][key], stride 40
    __shared__ alignas(16) bf16_t sP [4][16][32 + 8]; // per-wave P tile

    const int tid  = threadIdx.x;
    const int lane = tid & 31;
    const int wave = tid >> 5;
    const int half = lane >> 4;
    const int l16  = lane & 15;

    const int q0 = blockIdx.x * 64;
    const int b  = blockIdx.y / H;
    const int h  = blockIdx.y % H;

    // Q fragments, loaded once from global (A-matrix row m = l16)
    const int t_q = q0 + wave * 16 + l16;
    const bf16_t* qp = qk + (size_t)(b * T + t_q) * C2 + h * HD;
    v16bf aq[2];
#pragma unroll
    for (int ks = 0; ks < 2; ++ks) {
        v8bf lo = *(const v8bf*)(qp + ks * 32 + half * 8);
        v8bf hi = *(const v8bf*)(qp + ks * 32 + 16 + half * 8);
        aq[ks] = pack16(lo, hi);
    }

    v8f o[4];
#pragma unroll
    for (int nt = 0; nt < 4; ++nt) o[nt] = (v8f)0.0f;
    float mrow[8], lrow[8];
#pragma unroll
    for (int r = 0; r < 8; ++r) { mrow[r] = -1e30f; lrow[r] = 0.0f; }

    const float scale = 0.125f;  // 1/sqrt(64)
    const int qrow_base = q0 + wave * 16 + half * 8;
    const int nkb = (q0 + 64) / 32;
    const bf16_t* kbase = qk + (size_t)b * T * C2 + C + h * HD;      // + t*C2
    const bf16_t* vbase = vT + (size_t)(b * H + h) * HD * T;         // + hd*T

    for (int j = 0; j < nkb; ++j) {
        // stage K (32 x 128B) and Vt (64 x 64B): 256+256 x 16B units
#pragma unroll
        for (int i = 0; i < 2; ++i) {
            const int u = tid + i * 128;                   // 0..255
            { const int kk = u >> 3, seg = u & 7;
              copy16_g2l(&sK[kk][seg * 8],
                         kbase + (size_t)(j * 32 + kk) * C2 + seg * 8); }
            { const int hd = u >> 2, seg = u & 3;
              copy16_g2l(&sVt[hd][seg * 8],
                         vbase + (size_t)hd * T + j * 32 + seg * 8); }
        }
        copy_wait();
        __syncthreads();

        // S = Q @ K^T : 16 queries x 32 keys
        v8f s[2];
#pragma unroll
        for (int nt = 0; nt < 2; ++nt) {
            const int rowb = nt * 16 + l16;
            v8f sacc = (v8f)0.0f;
#pragma unroll
            for (int ks = 0; ks < 2; ++ks) {
                v8bf lo = *(const v8bf*)&sK[rowb][ks * 32 + half * 8];
                v8bf hi = *(const v8bf*)&sK[rowb][ks * 32 + 16 + half * 8];
                sacc = __builtin_amdgcn_wmma_f32_16x16x32_bf16(
                    false, aq[ks], false, pack16(lo, hi), (short)0, sacc,
                    false, false);
            }
            s[nt] = sacc;
        }

        // causal mask + scale
#pragma unroll
        for (int nt = 0; nt < 2; ++nt)
#pragma unroll
            for (int r = 0; r < 8; ++r) {
                const int qi = qrow_base + r;
                const int ki = j * 32 + nt * 16 + l16;
                const float v = s[nt][r] * scale;
                s[nt][r] = (ki <= qi) ? v : -1e30f;
            }

        // online softmax (row = 16-lane group)
#pragma unroll
        for (int r = 0; r < 8; ++r) {
            float m2 = fmaxf(s[0][r], s[1][r]);
            m2 = fmaxf(m2, __shfl_xor(m2, 1));
            m2 = fmaxf(m2, __shfl_xor(m2, 2));
            m2 = fmaxf(m2, __shfl_xor(m2, 4));
            m2 = fmaxf(m2, __shfl_xor(m2, 8));
            const float newm = fmaxf(mrow[r], m2);
            const float f  = __expf(mrow[r] - newm);
            const float p0 = __expf(s[0][r] - newm);
            const float p1 = __expf(s[1][r] - newm);
            s[0][r] = p0; s[1][r] = p1;
            float rs = p0 + p1;
            rs += __shfl_xor(rs, 1);
            rs += __shfl_xor(rs, 2);
            rs += __shfl_xor(rs, 4);
            rs += __shfl_xor(rs, 8);
            lrow[r] = lrow[r] * f + rs;
            mrow[r] = newm;
#pragma unroll
            for (int nt = 0; nt < 4; ++nt) o[nt][r] *= f;
        }

        // C-layout P -> wave-private LDS -> A-fragment (same-wave DS order)
#pragma unroll
        for (int nt = 0; nt < 2; ++nt)
#pragma unroll
            for (int r = 0; r < 8; ++r)
                sP[wave][half * 8 + r][nt * 16 + l16] = (bf16_t)s[nt][r];
        v8bf plo = *(const v8bf*)&sP[wave][l16][half * 8];
        v8bf phi = *(const v8bf*)&sP[wave][l16][16 + half * 8];
        v16bf ap = pack16(plo, phi);

        // O += P @ V
#pragma unroll
        for (int nt = 0; nt < 4; ++nt) {
            const int rowv = nt * 16 + l16;        // hd index
            v8bf lo = *(const v8bf*)&sVt[rowv][half * 8];
            v8bf hi = *(const v8bf*)&sVt[rowv][16 + half * 8];
            o[nt] = __builtin_amdgcn_wmma_f32_16x16x32_bf16(
                false, ap, false, pack16(lo, hi), (short)0, o[nt],
                false, false);
        }
        __syncthreads();
    }

    // normalize + write bf16 [B*T][C], col = h*HD + hd
#pragma unroll
    for (int r = 0; r < 8; ++r) {
        const float inv = 1.0f / lrow[r];
        const int t_o = q0 + wave * 16 + half * 8 + r;
        bf16_t* op = attout + (size_t)(b * T + t_o) * C + h * HD;
#pragma unroll
        for (int nt = 0; nt < 4; ++nt)
            op[nt * 16 + l16] = (bf16_t)(o[nt][r] * inv);
    }
}

// ---------------------------------------------------------------------------
extern "C" void kernel_launch(void* const* d_in, const int* in_sizes, int n_in,
                              void* d_out, int out_size, void* d_ws, size_t ws_size,
                              hipStream_t stream) {
    const float* x      = (const float*)d_in[0];   // [B,T,C]
    const float* w_qkv  = (const float*)d_in[1];   // [C,3C]
    const float* b_qkv  = (const float*)d_in[2];   // [3C]
    const float* w_proj = (const float*)d_in[3];   // [C,C]
    const float* b_proj = (const float*)d_in[4];   // [C]
    float* out = (float*)d_out;                    // [B,T,C]

    constexpr int B = CSA_B, T = CSA_T, C = CSA_C, H = CSA_H;
    constexpr int M  = B * T;       // 8192
    constexpr int N1 = 3 * C;       // 3072
    constexpr int K  = C;           // 1024

    // workspace layout (bf16 elements)
    bf16_t* xb     = (bf16_t*)d_ws;                     // M*K
    bf16_t* wqkvt  = xb     + (size_t)M * K;            // N1*K
    bf16_t* wprojt = wqkvt  + (size_t)N1 * K;           // C*K
    bf16_t* qk_ws  = wprojt + (size_t)C * K;            // M*2C
    bf16_t* vT_ws  = qk_ws  + (size_t)M * 2 * C;        // B*H*HD*T
    bf16_t* att_ws = vT_ws  + (size_t)B * C * T;        // M*C

    // 0) conversions / weight transposes (one-time, tiny)
    convert_f32_bf16_kernel<<<(size_t)M * K / 2048, 256, 0, stream>>>(x, xb);
    transpose_f32_bf16_kernel<<<dim3(N1 / 32, K / 32), dim3(32, 8), 0, stream>>>(
        w_qkv, wqkvt, K, N1);
    transpose_f32_bf16_kernel<<<dim3(C / 32, K / 32), dim3(32, 8), 0, stream>>>(
        w_proj, wprojt, K, C);

    // 1) QKV projection with split epilogue (q,k row-major; v transposed)
    gemm_bt_kernel<float, 1><<<dim3(N1 / 128, M / 128), 256, 0, stream>>>(
        xb, wqkvt, b_qkv, (float*)nullptr, qk_ws, vT_ws, M, N1, K);

    // 2) Flash attention
    attn_kernel<<<dim3(T / 64, B * H), 128, 0, stream>>>(qk_ws, vT_ws, att_ws);

    // 3) Output projection -> fp32
    gemm_bt_kernel<float, 0><<<dim3(C / 128, M / 128), 256, 0, stream>>>(
        att_ws, wprojt, b_proj, out, nullptr, nullptr, M, C, K);

    (void)in_sizes; (void)n_in; (void)out_size; (void)ws_size;
}
